// graph_encode_46196668236294
// MI455X (gfx1250) — matrix operand order repaired and checked by reading, not compiled, not run on hardware
//
#include <hip/hip_runtime.h>
#include <hip/hip_bf16.h>

// ---------------- problem constants ----------------
#define Bb   4
#define Ee   1536
#define Rr   512
#define Nn   2048           // E + R
#define HSZ  512
#define Hh   4
#define DK   128            // HSZ / H
#define FF   2048           // 4*HSZ
#define PROP 2
#define NEGV (-1.0e9f)
#define MROWS (Bb * Nn)     // 8192 rows of activations

typedef __bf16 bf16;
typedef __attribute__((ext_vector_type(16))) __bf16 v16bf;
typedef __attribute__((ext_vector_type(8)))  float  v8f;

// ---------------- WMMA helpers ----------------
__device__ __forceinline__ v8f wmma_bf16(v16bf a, v16bf b, v8f c) {
  return __builtin_amdgcn_wmma_f32_16x16x32_bf16(false, a, false, b, (short)0, c,
                                                 false, false);
}

// A fragment (16x32 bf16): lane<16 -> row = lane, k = {k0..k0+7, k0+16..k0+23}
//                          lane>=16 -> row = lane-16, k = {k0+8.., k0+24..}
__device__ __forceinline__ v16bf load_fragA(const bf16* __restrict__ base, int ldk,
                                            int row, int k0, int lane) {
  const bf16* p = base + (size_t)row * ldk + k0 + ((lane >> 4) << 3);
  v16bf r;
  ((uint4*)&r)[0] = *(const uint4*)p;
  ((uint4*)&r)[1] = *(const uint4*)(p + 16);
  return r;
}

// B fragment (32x16 bf16): lane<16 -> col = lane, k = {k0..k0+7, k0+16..k0+23}
//                          lane>=16 -> col = lane-16, k = {k0+8.., k0+24..}
// base is "column-major" source: row index = output column, contiguous K.
__device__ __forceinline__ v16bf load_fragB(const bf16* __restrict__ base, int ldk,
                                            int col, int k0, int lane) {
  const bf16* p = base + (size_t)col * ldk + k0 + ((lane >> 4) << 3);
  v16bf r;
  ((uint4*)&r)[0] = *(const uint4*)p;
  ((uint4*)&r)[1] = *(const uint4*)(p + 16);
  return r;
}

// ---------------- setup kernels ----------------
__global__ __launch_bounds__(256) void build_vg_kernel(
    const float* __restrict__ vents, const float* __restrict__ renc,
    const int* __restrict__ rels, bf16* __restrict__ vg) {
  size_t i = (size_t)blockIdx.x * 256 + threadIdx.x;
  if (i >= (size_t)MROWS * HSZ) return;
  int c = (int)(i % HSZ);
  size_t rn = i / HSZ;
  int n = (int)(rn % Nn);
  int b = (int)(rn / Nn);
  float v;
  if (n < Ee) {
    v = vents[((size_t)b * Ee + n) * HSZ + c];
  } else {
    int t = rels[b * Rr + (n - Ee)] & (Rr - 1);
    v = renc[(size_t)t * HSZ + c];
  }
  vg[i] = (bf16)v;
}

// W (K x Nc, f32 row-major) -> Wt (Nc x K, bf16)
__global__ __launch_bounds__(256) void wconv_kernel(
    const float* __restrict__ W, bf16* __restrict__ Wt, int K, int Nc) {
  size_t i = (size_t)blockIdx.x * 256 + threadIdx.x;
  if (i >= (size_t)K * Nc) return;
  int k = (int)(i % K);
  int n = (int)(i / K);
  Wt[i] = (bf16)W[(size_t)k * Nc + n];
}

__global__ __launch_bounds__(256) void copy_glob_kernel(
    const float* __restrict__ gents, float* __restrict__ glob) {
  int i = blockIdx.x * 256 + threadIdx.x;
  if (i >= Bb * HSZ) return;
  int b = i / HSZ, c = i % HSZ;
  glob[i] = gents[((size_t)b * Nn + Ee) * HSZ + c];
}

// ---------------- generic WMMA GEMM ----------------
// C[M,Nc] = A[M,K](bf16 row-major) * Bt[Nc,K](bf16) + bias, optional PReLU.
// Outputs: Cf (f32 row-major), Cb (bf16 row-major), Ct (bf16 transposed: [Nc][M]).
__global__ __launch_bounds__(128) void gemm_kernel(
    const bf16* __restrict__ A, const bf16* __restrict__ Bt,
    const float* __restrict__ bias,
    float* __restrict__ Cf, bf16* __restrict__ Cb, bf16* __restrict__ Ct,
    const float* __restrict__ prelu, int M, int K, int Nc) {
  int lane = threadIdx.x & 31;
  int wave = threadIdx.x >> 5;
  int tileM = (blockIdx.x * 4 + wave) * 32;
  int tileN = blockIdx.y * 32;
  if (tileM >= M || tileN >= Nc) return;

  v8f c00 = {}, c01 = {}, c10 = {}, c11 = {};
  int rA0 = tileM + (lane & 15);
  int rA1 = rA0 + 16;
  int cB0 = tileN + (lane & 15);
  int cB1 = cB0 + 16;

  for (int k0 = 0; k0 < K; k0 += 32) {
    v16bf a0 = load_fragA(A, K, rA0, k0, lane);
    v16bf a1 = load_fragA(A, K, rA1, k0, lane);
    v16bf b0 = load_fragB(Bt, K, cB0, k0, lane);
    v16bf b1 = load_fragB(Bt, K, cB1, k0, lane);
    c00 = wmma_bf16(a0, b0, c00);
    c01 = wmma_bf16(a0, b1, c01);
    c10 = wmma_bf16(a1, b0, c10);
    c11 = wmma_bf16(a1, b1, c11);
  }

  // epilogue: C layout -> lane holds column (lane&15)+16*ni, vgpr r holds row
  // r + 8*(lane>=16) + 16*mi
  v8f ct[2][2];
  ct[0][0] = c00; ct[0][1] = c01; ct[1][0] = c10; ct[1][1] = c11;
  int colLo = tileN + (lane & 15);
  int rowLo = tileM + ((lane >> 4) << 3);
#pragma unroll
  for (int mi = 0; mi < 2; ++mi) {
#pragma unroll
    for (int ni = 0; ni < 2; ++ni) {
      int col = colLo + ni * 16;
      int rbase = rowLo + mi * 16;
      float bia = bias ? bias[col] : 0.0f;
      float pa = prelu ? prelu[col] : 0.0f;
#pragma unroll
      for (int r = 0; r < 8; ++r) {
        float v = ct[mi][ni][r] + bia;
        if (prelu) v = (v > 0.0f) ? v : pa * v;
        size_t idx = (size_t)(rbase + r) * Nc + col;
        if (Cf) Cf[idx] = v;
        if (Cb) Cb[idx] = (bf16)v;
        if (Ct) Ct[(size_t)col * M + (rbase + r)] = (bf16)v;
      }
    }
  }
}

// ---------------- flash attention ----------------
// One wave handles 16 query rows for one (b, h). Computes S^T = K * Q^T so the
// exp'd probabilities land exactly in A-fragment layout for the P*V WMMA.
__global__ __launch_bounds__(128) void attn_kernel(
    const bf16* __restrict__ q, const bf16* __restrict__ k,
    const bf16* __restrict__ vt, const int* __restrict__ adj,
    bf16* __restrict__ o) {
  int lane = threadIdx.x & 31;
  int w = blockIdx.x * 4 + (threadIdx.x >> 5);
  if (w >= Bb * Hh * (Nn / 16)) return;
  int qt = w & 127;          // N/16 = 128 query tiles
  int h = (w >> 7) & 3;
  int b = w >> 9;
  int n0 = qt * 16;

  const bf16* qbase = q + (size_t)b * Nn * HSZ;
  const bf16* kbase = k + (size_t)b * Nn * HSZ;
  int qcol = n0 + (lane & 15);
  const int* adjrow = adj + ((size_t)b * Nn + qcol) * Nn;
  int hiSel = ((lane >> 4) << 3);   // 0 for lanes 0-15, 8 for lanes 16-31

  v16bf qf[4];
#pragma unroll
  for (int c = 0; c < 4; ++c)
    qf[c] = load_fragB(qbase, HSZ, qcol, h * DK + c * 32, lane);

  v8f acc[8];
#pragma unroll
  for (int c = 0; c < 8; ++c) acc[c] = (v8f){};
  float m_run = -3.0e38f;
  float l_part = 0.0f;
  const float scale = 0.08838834764831845f;  // 1/sqrt(128)

  for (int m0 = 0; m0 < Nn; m0 += 32) {
    v8f s0 = {}, s1 = {};
#pragma unroll
    for (int c = 0; c < 4; ++c) {
      int dk0 = h * DK + c * 32;
      v16bf kf0 = load_fragA(kbase, HSZ, m0 + (lane & 15), dk0, lane);
      v16bf kf1 = load_fragA(kbase, HSZ, m0 + 16 + (lane & 15), dk0, lane);
      s0 = wmma_bf16(kf0, qf[c], s0);
      s1 = wmma_bf16(kf1, qf[c], s1);
    }
    // scale + mask; S^T layout: vgpr r = key row m0 + r + hiSel (+16 for s1),
    // lane = query column qcol.
    float sv[16];
    int mrow = m0 + hiSel;
#pragma unroll
    for (int r = 0; r < 8; ++r) {
      float x = s0[r] * scale;
      if (adjrow[mrow + r] == 0) x = NEGV;
      sv[r] = x;
    }
#pragma unroll
    for (int r = 0; r < 8; ++r) {
      float x = s1[r] * scale;
      if (adjrow[mrow + 16 + r] == 0) x = NEGV;
      sv[8 + r] = x;
    }
    // online softmax (per query = per lane pair L / L+16)
    float mx = sv[0];
#pragma unroll
    for (int i = 1; i < 16; ++i) mx = fmaxf(mx, sv[i]);
    mx = fmaxf(mx, __shfl_xor(mx, 16));
    float m_new = fmaxf(m_run, mx);
    float alpha = __expf(m_run - m_new);
    m_run = m_new;

    v16bf pfrag;
    float ps = 0.0f;
#pragma unroll
    for (int i = 0; i < 16; ++i) {
      float p = __expf(sv[i] - m_new);
      ps += p;
      pfrag[i] = (bf16)p;
    }
    l_part = l_part * alpha + ps;

    // rescale accumulators: alpha is per-query; O layout has query rows in
    // vgprs -> broadcast via shfl.
    float ar[8];
#pragma unroll
    for (int r = 0; r < 8; ++r) ar[r] = __shfl(alpha, r + hiSel, 32);
#pragma unroll
    for (int c = 0; c < 8; ++c)
#pragma unroll
      for (int r = 0; r < 8; ++r) acc[c][r] *= ar[r];

    // O += P(16x32) * V(32x16 per dk chunk); V read from transposed vt.
#pragma unroll
    for (int c = 0; c < 8; ++c) {
      const bf16* p = vt + (size_t)(h * DK + c * 16 + (lane & 15)) * MROWS +
                      (size_t)b * Nn + m0 + ((lane >> 4) << 3);
      v16bf vb;
      ((uint4*)&vb)[0] = *(const uint4*)p;
      ((uint4*)&vb)[1] = *(const uint4*)(p + 16);
      acc[c] = wmma_bf16(pfrag, vb, acc[c]);
    }
  }

  // finalize: combine row-sum halves, divide, store bf16 row-major o.
  float l = l_part + __shfl_xor(l_part, 16);
  float lr[8];
#pragma unroll
  for (int r = 0; r < 8; ++r) lr[r] = 1.0f / __shfl(l, r + hiSel, 32);
#pragma unroll
  for (int c = 0; c < 8; ++c) {
    int col = h * DK + c * 16 + (lane & 15);
#pragma unroll
    for (int r = 0; r < 8; ++r) {
      int row = n0 + r + hiSel;
      o[((size_t)b * Nn + row) * HSZ + col] = (bf16)(acc[c][r] * lr[r]);
    }
  }
}

// ---------------- LayerNorm (cols == HSZ == 512), optional residual ----------
__global__ __launch_bounds__(256) void ln_kernel(
    const float* __restrict__ x, const float* __restrict__ res,
    const float* __restrict__ g, const float* __restrict__ bb,
    float* __restrict__ outf, bf16* __restrict__ outb) {
  __shared__ float red[16];
  int row = blockIdx.x;
  int t = threadIdx.x;
  const float* xr = x + (size_t)row * HSZ;
  float v0 = xr[t], v1 = xr[t + 256];
  if (res) {
    const float* rr = res + (size_t)row * HSZ;
    v0 += rr[t];
    v1 += rr[t + 256];
  }
  float s = v0 + v1;
  float ss = v0 * v0 + v1 * v1;
#pragma unroll
  for (int off = 16; off > 0; off >>= 1) {
    s += __shfl_xor(s, off, 32);
    ss += __shfl_xor(ss, off, 32);
  }
  int lane = t & 31, wv = t >> 5;
  if (lane == 0) { red[wv] = s; red[8 + wv] = ss; }
  __syncthreads();
  if (wv == 0) {
    float a = (lane < 8) ? red[lane] : 0.0f;
    float c2 = (lane < 8) ? red[8 + lane] : 0.0f;
#pragma unroll
    for (int off = 4; off > 0; off >>= 1) {
      a += __shfl_xor(a, off, 32);
      c2 += __shfl_xor(c2, off, 32);
    }
    if (lane == 0) { red[0] = a; red[8] = c2; }
  }
  __syncthreads();
  float mu = red[0] * (1.0f / HSZ);
  float var = red[8] * (1.0f / HSZ) - mu * mu;
  float rstd = rsqrtf(var + 1e-5f);
  float y0 = (v0 - mu) * rstd * g[t] + bb[t];
  float y1 = (v1 - mu) * rstd * g[t + 256] + bb[t + 256];
  size_t base = (size_t)row * HSZ;
  if (outf) { outf[base + t] = y0; outf[base + t + 256] = y1; }
  if (outb) { outb[base + t] = (bf16)y0; outb[base + t + 256] = (bf16)y1; }
}

// ---------------- host orchestration ----------------
extern "C" void kernel_launch(void* const* d_in, const int* in_sizes, int n_in,
                              void* d_out, int out_size, void* d_ws, size_t ws_size,
                              hipStream_t stream) {
  (void)in_sizes; (void)n_in; (void)out_size; (void)ws_size;
  const int*   adj    = (const int*)d_in[0];
  const int*   rels   = (const int*)d_in[1];
  const float* vents  = (const float*)d_in[2];
  const float* renc_w = (const float*)d_in[3];
  const float* Wq = (const float*)d_in[4];
  const float* bq = (const float*)d_in[5];
  const float* Wk = (const float*)d_in[6];
  const float* bk = (const float*)d_in[7];
  const float* Wv = (const float*)d_in[8];
  const float* bv = (const float*)d_in[9];
  const float* Wo = (const float*)d_in[10];
  const float* bo = (const float*)d_in[11];
  const float* l1w = (const float*)d_in[12];
  const float* l1b = (const float*)d_in[13];
  const float* l2w = (const float*)d_in[14];
  const float* l2b = (const float*)d_in[15];
  const float* prA = (const float*)d_in[16];
  const float* lng = (const float*)d_in[17];
  const float* lnb = (const float*)d_in[18];

  float* glob  = (float*)d_out;                 // (B, HSZ)
  float* gents = (float*)d_out + Bb * HSZ;      // (B, N, HSZ) — doubles as final vg f32

  const size_t MB = 1u << 20;
  char* ws = (char*)d_ws;
  bf16*  vg_b  = (bf16*)(ws + 0 * MB);     //  8 MB  (B*N*HSZ)
  bf16*  q_b   = (bf16*)(ws + 8 * MB);     //  8 MB
  bf16*  k_b   = (bf16*)(ws + 16 * MB);    //  8 MB
  bf16*  vt_b  = (bf16*)(ws + 24 * MB);    //  8 MB  transposed: [HSZ][B*N]
  bf16*  o_b   = (bf16*)(ws + 32 * MB);    //  8 MB
  bf16*  t_b   = (bf16*)(ws + 40 * MB);    //  8 MB
  bf16*  h_b   = (bf16*)(ws + 48 * MB);    // 32 MB  (B*N*FF)
  float* tmp_f = (float*)(ws + 80 * MB);   // 16 MB  (B*N*HSZ)
  float* t_f   = (float*)(ws + 96 * MB);   // 16 MB
  bf16*  wts   = (bf16*)(ws + 112 * MB);   // 12 MB  transposed bf16 weights

  // per-layer transposed weight pointers
  const size_t WSQ = (size_t)HSZ * HSZ;     // 262144 elems
  const size_t WSF = (size_t)HSZ * FF;      // 1048576 elems
  bf16* wqT[PROP]; bf16* wkT[PROP]; bf16* wvT[PROP]; bf16* woT[PROP];
  bf16* l1T[PROP]; bf16* l2T[PROP];
  {
    bf16* p = wts;
    for (int j = 0; j < PROP; ++j) {
      wqT[j] = p; p += WSQ;
      wkT[j] = p; p += WSQ;
      wvT[j] = p; p += WSQ;
      woT[j] = p; p += WSQ;
      l1T[j] = p; p += WSF;
      l2T[j] = p; p += WSF;
    }
  }

  // 1) build vg (bf16)
  {
    int nblk = (int)(((size_t)MROWS * HSZ + 255) / 256);
    build_vg_kernel<<<nblk, 256, 0, stream>>>(vents, renc_w, rels, vg_b);
  }
  // 2) convert weights (both layers)
  for (int j = 0; j < PROP; ++j) {
    int nsq = (int)((WSQ + 255) / 256), nff = (int)((WSF + 255) / 256);
    wconv_kernel<<<nsq, 256, 0, stream>>>(Wq + j * WSQ, wqT[j], HSZ, HSZ);
    wconv_kernel<<<nsq, 256, 0, stream>>>(Wk + j * WSQ, wkT[j], HSZ, HSZ);
    wconv_kernel<<<nsq, 256, 0, stream>>>(Wv + j * WSQ, wvT[j], HSZ, HSZ);
    wconv_kernel<<<nsq, 256, 0, stream>>>(Wo + j * WSQ, woT[j], HSZ, HSZ);
    wconv_kernel<<<nff, 256, 0, stream>>>(l1w + j * WSF, l1T[j], HSZ, FF);
    wconv_kernel<<<nff, 256, 0, stream>>>(l2w + j * WSF, l2T[j], FF, HSZ);
  }

  dim3 gQ(MROWS / 128, HSZ / 32);   // GEMMs with Nc=512
  dim3 gF(MROWS / 128, FF / 32);    // FFN up with Nc=2048

  for (int j = 0; j < PROP; ++j) {
    // Q, K projections (bf16 out)
    gemm_kernel<<<gQ, 128, 0, stream>>>(vg_b, wqT[j], bq + j * HSZ,
                                        nullptr, q_b, nullptr, nullptr,
                                        MROWS, HSZ, HSZ);
    gemm_kernel<<<gQ, 128, 0, stream>>>(vg_b, wkT[j], bk + j * HSZ,
                                        nullptr, k_b, nullptr, nullptr,
                                        MROWS, HSZ, HSZ);
    // V projection -> transposed bf16 output for attention PV GEMM
    gemm_kernel<<<gQ, 128, 0, stream>>>(vg_b, wvT[j], bv + j * HSZ,
                                        nullptr, nullptr, vt_b, nullptr,
                                        MROWS, HSZ, HSZ);
    // masked flash attention -> o (bf16)
    attn_kernel<<<(Bb * Hh * (Nn / 16)) / 4, 128, 0, stream>>>(q_b, k_b, vt_b,
                                                               adj, o_b);
    // output projection -> f32
    gemm_kernel<<<gQ, 128, 0, stream>>>(o_b, woT[j], bo + j * HSZ,
                                        tmp_f, nullptr, nullptr, nullptr,
                                        MROWS, HSZ, HSZ);
    // t = LN(o)
    ln_kernel<<<MROWS, 256, 0, stream>>>(tmp_f, nullptr, lng + j * HSZ,
                                         lnb + j * HSZ, t_f, t_b);
    // h = PReLU(t @ l1 + b1) -> bf16
    gemm_kernel<<<gF, 128, 0, stream>>>(t_b, l1T[j], l1b + j * FF,
                                        nullptr, h_b, nullptr, prA + j * FF,
                                        MROWS, HSZ, FF);
    // f = h @ l2 + b2 -> f32
    gemm_kernel<<<gQ, 128, 0, stream>>>(h_b, l2T[j], l2b + j * HSZ,
                                        tmp_f, nullptr, nullptr, nullptr,
                                        MROWS, FF, HSZ);
    // vg = LN(f + t) -> gents (f32) and vg_b (bf16) for next layer
    ln_kernel<<<MROWS, 256, 0, stream>>>(tmp_f, t_f, lng + j * HSZ,
                                         lnb + j * HSZ, gents, vg_b);
  }

  // glob = vg[:, E]
  copy_glob_kernel<<<(Bb * HSZ + 255) / 256, 256, 0, stream>>>(gents, glob);
}
